// BiNet_30279519437063
// MI455X (gfx1250) — matrix operand order repaired and checked by reading, not compiled, not run on hardware
//
#include <hip/hip_runtime.h>
#include <hip/hip_bf16.h>
#include <math.h>

typedef _Float16 h16 __attribute__((ext_vector_type(16)));
typedef float    f8  __attribute__((ext_vector_type(8)));

#define N_BATCH 2048

// ---------------------------------------------------------------------------
// Prep: per-layer weight scales (mean|w| per out-channel), sign-pack conv
// weights for layers 2..7 into WMMA A-fragment layout, layer-1 sign weights,
// and zero the BN-stats accumulators (fresh every launch).
// A-fragment (16x32 f16, wave32), k contiguous per lane half:
//   lane<16 : M=lane,    elems 0..7 -> K 0..7,  elems 8..15 -> K 8..15
//   lane>=16: M=lane-16, elems 0..7 -> K 16..23, elems 8..15 -> K 24..31
// ---------------------------------------------------------------------------
__global__ void prep_kernel(const float* w1, const float* w2, const float* w3,
                            const float* w4, const float* w5, const float* w6,
                            const float* w7, _Float16* wpack, float* scales,
                            float* w1sign, float* stats)
{
    const int L   = blockIdx.x;      // 0..6 = layer1..7, 7 = zero stats
    const int tid = threadIdx.x;
    const int bd  = blockDim.x;
    if (L == 7) {
        for (int i = tid; i < 7 * 160; i += bd) stats[i] = 0.f;
        return;
    }
    const float* W[7] = {w1, w2, w3, w4, w5, w6, w7};
    const int CIN[7]  = {1, 8, 12, 32, 64, 64, 64};
    const int COUT[7] = {8, 12, 32, 64, 64, 64, 72};
    const int KK[7]   = {16, 12, 9, 7, 5, 3, 3};
    const int MT[7]   = {1, 1, 2, 4, 4, 4, 5};
    const int KS[7]   = {1, 3, 4, 7, 10, 6, 6};
    const int WOFF[7] = {0, 0, 1536, 5632, 19968, 40448, 52736}; // in halves

    const int cin  = CIN[L], cout = COUT[L], k = KK[L];
    const int kc   = cin * k;
    const float* w = W[L];

    // per-out-channel scale = mean|w|
    for (int c = tid; c < cout; c += bd) {
        float s = 0.f;
        for (int t = 0; t < kc; ++t) s += fabsf(w[c * kc + t]);
        scales[L * 80 + c] = s / (float)kc;
    }

    if (L == 0) { // layer 1: sign weights as f32 (VALU path)
        for (int i = tid; i < 8 * 16; i += bd) {
            float v = w[i];
            w1sign[i] = (float)((v > 0.f) - (v < 0.f));
        }
        return;
    }

    const int mt = MT[L], ks = KS[L], Ktot = kc;
    const int total = mt * ks * 32 * 16;
    _Float16* dst = wpack + WOFF[L];
    for (int i = tid; i < total; i += bd) {
        int e     = i & 15;
        int lane  = (i >> 4) & 31;
        int rest  = i >> 9;            // mtile*ks + kstep
        int kstep = rest % ks;
        int mtile = rest / ks;
        int m  = mtile * 16 + (lane & 15);
        int kq = ((lane >= 16) ? 2 : 0) + ((e < 8) ? 0 : 1);
        int K  = kstep * 32 + kq * 8 + (e & 7);
        _Float16 v = (_Float16)0.f;
        if (m < cout && K < Ktot) {
            float x = w[m * kc + K];   // K flattens (c_in, kk) as c_in*k + kk
            v = (_Float16)(float)((x > 0.f) - (x < 0.f));
        }
        dst[i] = v;
    }
}

// ---------------------------------------------------------------------------
// Layer 1: f32 conv(k=16,s=2,p=7) fused with maxpool(8,4) + BN partial stats.
// One block per batch sample; x row staged in LDS.
// ---------------------------------------------------------------------------
__global__ void conv_pool1_kernel(const float* __restrict__ x,
                                  const float* __restrict__ w1sign,
                                  const float* __restrict__ scales,
                                  float* __restrict__ y_pool,
                                  float* __restrict__ stats)
{
    __shared__ float xs[3600];
    __shared__ float wsh[128];
    __shared__ float ssum[8], ssq[8];
    const int b = blockIdx.x, tid = threadIdx.x, bd = blockDim.x;
    for (int i = tid; i < 3600; i += bd) xs[i] = x[(size_t)b * 3600 + i];
    for (int i = tid; i < 128; i += bd)  wsh[i] = w1sign[i];
    if (tid < 8) { ssum[tid] = 0.f; ssq[tid] = 0.f; }
    __syncthreads();
    for (int i = tid; i < 8 * 449; i += bd) {
        int c = i / 449, lp = i - c * 449;
        float best = -3.0e38f;
        for (int j = 0; j < 8; ++j) {          // pool window (k=8, s=4)
            int pos = lp * 4 + j;              // conv position < 1800
            int g0  = pos * 2 - 7;
            float s = 0.f;
            #pragma unroll
            for (int t = 0; t < 16; ++t) {
                int g = g0 + t;
                float xv = (g >= 0 && g < 3600) ? xs[g] : 0.f;
                s += wsh[c * 16 + t] * xv;
            }
            best = fmaxf(best, s);
        }
        float val = best * scales[c];          // scale > 0: commutes with max
        y_pool[((size_t)b * 8 + c) * 449 + lp] = val;
        atomicAdd(&ssum[c], val);
        atomicAdd(&ssq[c], val * val);
    }
    __syncthreads();
    if (tid < 8) {
        atomicAdd(&stats[tid], ssum[tid]);
        atomicAdd(&stats[80 + tid], ssq[tid]);
    }
}

// ---------------------------------------------------------------------------
// Layers 2..7: implicit-GEMM conv via v_wmma_f32_16x16x32_f16, fully
// specialized per layer: the k-step loop unrolls into a static chain of
// WMMAs and all LDS gather offsets fold to constants.
// B-fragment (32x16 f16): lane<16: N=lane, K 0..15; lane>=16: N=lane-16, K 16..31.
// D (16x16 f32, 8 VGPRs): elem r -> M = r + 8*(lane>=16), N = lane&15.
// ---------------------------------------------------------------------------
template <int CIN, int COUT, int KW, int STRIDE, int PAD,
          int LIN, int LC, int KSTEPS, int KTOT, int WN>
__global__ void conv_wmma_t(const _Float16* __restrict__ act,
                            const _Float16* __restrict__ wpack,
                            const float* __restrict__ scales,
                            float* __restrict__ y_conv)
{
    __shared__ _Float16 slds[CIN * WN + 32];
    const int b     = blockIdx.x;
    const int ptile = blockIdx.y;
    const int tid   = threadIdx.x;
    const int wave  = tid >> 5;                // = mtile
    const int lane  = tid & 31;
    const int laneN = lane & 15;
    const int hi    = lane >> 4;
    const int base  = ptile * 16 * STRIDE - PAD;
    for (int i = tid; i < CIN * WN; i += blockDim.x) {
        int c = i / WN, off = i - c * WN;
        int g = base + off;
        slds[i] = (g >= 0 && g < LIN)
                    ? act[((size_t)b * CIN + c) * LIN + g] : (_Float16)0.f;
    }
    __syncthreads();

    f8 acc = {};
    #pragma unroll
    for (int ks = 0; ks < KSTEPS; ++ks) {
        h16 afrag = *(const h16*)(wpack + ((wave * KSTEPS + ks) * 32 + lane) * 16);
        h16 bfrag;
        const int kbase = ks * 32 + hi * 16;   // hi is the only runtime term
        #pragma unroll
        for (int e = 0; e < 16; ++e) {
            // K = kbase + e ; split into the two half-wave variants so the
            // channel/tap decomposition is compile-time in both.
            int Klo = ks * 32 + e;             // lanes 0..15
            int Khi = ks * 32 + 16 + e;        // lanes 16..31
            _Float16 vlo = (_Float16)0.f, vhi = (_Float16)0.f;
            if (Klo < KTOT)
                vlo = slds[(Klo / KW) * WN + laneN * STRIDE + (Klo % KW)];
            if (Khi < KTOT)
                vhi = slds[(Khi / KW) * WN + laneN * STRIDE + (Khi % KW)];
            bfrag[e] = hi ? vhi : vlo;
        }
        (void)kbase;
        acc = __builtin_amdgcn_wmma_f32_16x16x32_f16(
                  false, afrag, false, bfrag, (short)0, acc, false, false);
    }

    int pos = ptile * 16 + laneN;
    if (pos < LC) {
        #pragma unroll
        for (int r = 0; r < 8; ++r) {
            int m = wave * 16 + r + hi * 8;
            if (m < COUT)
                y_conv[((size_t)b * COUT + m) * LC + pos] = acc[r] * scales[m];
        }
    }
}

// ---------------------------------------------------------------------------
// Maxpool + per-channel BN partial statistics (LDS reduce -> global atomics).
// One block per batch sample.
// ---------------------------------------------------------------------------
__global__ void pool_stats_kernel(const float* __restrict__ y_conv,
                                  float* __restrict__ y_pool,
                                  float* __restrict__ stats,
                                  int C, int L_c, int L_p, int pk, int ps)
{
    __shared__ float ssum[80], ssq[80];
    const int b = blockIdx.x, tid = threadIdx.x, bd = blockDim.x;
    for (int c = tid; c < C; c += bd) { ssum[c] = 0.f; ssq[c] = 0.f; }
    __syncthreads();
    const int total = C * L_p;
    for (int i = tid; i < total; i += bd) {
        int c = i / L_p, lp = i - c * L_p;
        const float* src = y_conv + ((size_t)b * C + c) * L_c + lp * ps;
        float m = -3.0e38f;
        for (int j = 0; j < pk; ++j) m = fmaxf(m, src[j]);
        y_pool[((size_t)b * C + c) * L_p + lp] = m;
        atomicAdd(&ssum[c], m);
        atomicAdd(&ssq[c], m * m);
    }
    __syncthreads();
    for (int c = tid; c < C; c += bd) {
        atomicAdd(&stats[c], ssum[c]);
        atomicAdd(&stats[80 + c], ssq[c]);
    }
}

// ---------------------------------------------------------------------------
// BatchNorm (batch stats, biased var) + sign() -> f16 {-1,0,+1} activations.
// ---------------------------------------------------------------------------
__global__ void binact_kernel(const float* __restrict__ y_pool,
                              const float* __restrict__ stats,
                              const float* __restrict__ gamma,
                              const float* __restrict__ beta,
                              _Float16* __restrict__ act,
                              int C, int L_p, float inv_count, int total)
{
    int idx = blockIdx.x * blockDim.x + threadIdx.x;
    if (idx >= total) return;
    int c = (idx / L_p) % C;
    float mean = stats[c] * inv_count;
    float var  = fmaxf(stats[80 + c] * inv_count - mean * mean, 0.f);
    float xv   = y_pool[idx];
    float v    = gamma[c] * (xv - mean) * rsqrtf(var + 1e-5f) + beta[c];
    act[idx]   = (_Float16)(float)((v > 0.f) - (v < 0.f));
}

// ---------------------------------------------------------------------------
// FC: [2048,216] x sign(wfc)^T * mean|wfc_row|  ->  [2048,5]
// ---------------------------------------------------------------------------
__global__ void fc_kernel(const _Float16* __restrict__ act,
                          const float* __restrict__ wfc,
                          float* __restrict__ out)
{
    int idx = blockIdx.x * blockDim.x + threadIdx.x;
    if (idx >= N_BATCH * 5) return;
    int b = idx / 5, cls = idx - b * 5;
    const float* wr = wfc + cls * 216;
    float sc = 0.f;
    for (int t = 0; t < 216; ++t) sc += fabsf(wr[t]);
    sc *= (1.f / 216.f);
    const _Float16* a = act + (size_t)b * 216;   // flatten order = c*3 + l
    float s = 0.f;
    for (int t = 0; t < 216; ++t) {
        float w = wr[t];
        s += (float)a[t] * (float)((w > 0.f) - (w < 0.f));
    }
    out[idx] = s * sc;
}

// ---------------------------------------------------------------------------
extern "C" void kernel_launch(void* const* d_in, const int* in_sizes, int n_in,
                              void* d_out, int out_size, void* d_ws, size_t ws_size,
                              hipStream_t stream)
{
    (void)in_sizes; (void)n_in; (void)out_size; (void)ws_size;
    // dict order: x, (w1,g1,b1), (w2,g2,b2), ..., (w7,g7,b7), wfc
    const float* x = (const float*)d_in[0];
    const float *w[7], *g[7], *bb[7];
    for (int i = 0; i < 7; ++i) {
        w[i]  = (const float*)d_in[1 + 3 * i];
        g[i]  = (const float*)d_in[2 + 3 * i];
        bb[i] = (const float*)d_in[3 + 3 * i];
    }
    const float* wfc = (const float*)d_in[22];
    float* out = (float*)d_out;

    // workspace layout (256B-aligned regions)
    char* base = (char*)d_ws;
    _Float16* wpack  = (_Float16*)(base + 0);                 // <=136.2 KB
    float*    scales = (float*)(base + 262144);               // 7*80 f32
    float*    w1sign = (float*)(base + 266240);               // 128 f32
    float*    stats  = (float*)(base + 270336);               // 7*160 f32
    const size_t MB = 1048576;
    _Float16* act0   = (_Float16*)(base + 1 * MB);            // 16 MB
    _Float16* act1   = (_Float16*)(base + 17 * MB);           // 16 MB
    float*    convb  = (float*)(base + 33 * MB);              // 32 MB
    float*    poolb  = (float*)(base + 65 * MB);              // 32 MB

    prep_kernel<<<8, 256, 0, stream>>>(w[0], w[1], w[2], w[3], w[4], w[5], w[6],
                                       wpack, scales, w1sign, stats);

    // ---- layer 1 (VALU f32, fused conv+pool) ----
    conv_pool1_kernel<<<N_BATCH, 256, 0, stream>>>(x, w1sign, scales, poolb, stats);
    {
        const int C = 8, Lp = 449, total = N_BATCH * C * Lp;
        binact_kernel<<<(total + 255) / 256, 256, 0, stream>>>(
            poolb, stats, g[0], bb[0], act0, C, Lp,
            1.f / (float)(N_BATCH * Lp), total);
    }

    // ---- layers 2..7 (WMMA implicit GEMM, fully specialized) ----
    const int COUT[6]  = {12, 32, 64, 64, 64, 72};
    const int LC[6]    = {224, 111, 54, 26, 13, 6};
    const int PK[6]    = {4, 5, 4, 2, 2, 2};
    const int PS[6]    = {2, 2, 2, 2, 2, 2};
    const int LP[6]    = {111, 54, 26, 13, 6, 3};
    const int WOFF[6]  = {0, 1536, 5632, 19968, 40448, 52736};

    _Float16* acts[2] = {act0, act1};
    for (int i = 0; i < 6; ++i) {
        _Float16* ain  = acts[i & 1];
        _Float16* aout = acts[(i + 1) & 1];
        float* statL   = stats + (i + 1) * 160;
        const _Float16* wp = wpack + WOFF[i];
        const float*    sc = scales + (i + 1) * 80;

        switch (i) {
        case 0: //        CIN CO  K  S  P  LIN  LC  KS KTOT WN
            conv_wmma_t<  8, 12,12, 2, 5, 449, 224, 3,  96, 42>
                <<<dim3(N_BATCH, 14), 32 * 1, 0, stream>>>(ain, wp, sc, convb);
            break;
        case 1:
            conv_wmma_t< 12, 32, 9, 1, 4, 111, 111, 4, 108, 24>
                <<<dim3(N_BATCH, 7), 32 * 2, 0, stream>>>(ain, wp, sc, convb);
            break;
        case 2:
            conv_wmma_t< 32, 64, 7, 1, 3,  54,  54, 7, 224, 22>
                <<<dim3(N_BATCH, 4), 32 * 4, 0, stream>>>(ain, wp, sc, convb);
            break;
        case 3:
            conv_wmma_t< 64, 64, 5, 1, 2,  26,  26,10, 320, 20>
                <<<dim3(N_BATCH, 2), 32 * 4, 0, stream>>>(ain, wp, sc, convb);
            break;
        case 4:
            conv_wmma_t< 64, 64, 3, 1, 1,  13,  13, 6, 192, 18>
                <<<dim3(N_BATCH, 1), 32 * 4, 0, stream>>>(ain, wp, sc, convb);
            break;
        case 5:
            conv_wmma_t< 64, 72, 3, 1, 1,   6,   6, 6, 192, 18>
                <<<dim3(N_BATCH, 1), 32 * 5, 0, stream>>>(ain, wp, sc, convb);
            break;
        }

        pool_stats_kernel<<<N_BATCH, 256, 0, stream>>>(
            convb, poolb, statL, COUT[i], LC[i], LP[i], PK[i], PS[i]);
        const int total = N_BATCH * COUT[i] * LP[i];
        binact_kernel<<<(total + 255) / 256, 256, 0, stream>>>(
            poolb, statL, g[i + 1], bb[i + 1], aout,
            COUT[i], LP[i], 1.f / (float)(N_BATCH * LP[i]), total);
    }

    // final activations land in acts[0] after 6 ping-pongs
    fc_kernel<<<(N_BATCH * 5 + 255) / 256, 256, 0, stream>>>(acts[0], wfc, out);
}
